// MultiHeadAttention_70789650973287
// MI455X (gfx1250) — compile-verified, matrix operand order checked by reading, not run on hardware
//
#include <hip/hip_runtime.h>
#include <hip/hip_bf16.h>

// ---------------- problem constants ----------------
#define B_   2
#define S_   2048
#define D_   1024
#define H_   16
#define DH_  64
#define HB_  32            // H*B
#define M_   4096          // B*S
#define TEMP_INV 0.125f    // 1/sqrt(DH)
#define EPS_ 1e-5f

// ---------------- vector types ----------------
typedef __bf16 bf16_t;
typedef __attribute__((ext_vector_type(4)))  float    v4f;
typedef __attribute__((ext_vector_type(8)))  float    v8f;
typedef __attribute__((ext_vector_type(8)))  bf16_t   v8bf;
typedef __attribute__((ext_vector_type(16))) bf16_t   v16bf;
typedef __attribute__((ext_vector_type(4)))  unsigned uint4v;
typedef __attribute__((ext_vector_type(8)))  unsigned uint8v;

// ---------------- helpers ----------------
static __device__ __forceinline__ unsigned short f2bf_bits(float f) {
    unsigned u = __float_as_uint(f);
    return (unsigned short)((u + 0x7FFFu + ((u >> 16) & 1u)) >> 16);  // RNE
}
static __device__ __forceinline__ bf16_t f2bf(float f) {
    return __builtin_bit_cast(bf16_t, f2bf_bits(f));
}

// WMMA wrapper: D = A(16x32 bf16) * B(32x16 bf16) + C(16x16 f32)
static __device__ __forceinline__ v8f wmma_bf16(v16bf a, v16bf b, v8f c) {
    return __builtin_amdgcn_wmma_f32_16x16x32_bf16(false, a, false, b,
                                                   (short)0, c, false, false);
}

// Load a 16x32 A/B fragment whose K-runs are contiguous at p[0..7] and p[16..23].
// Caller points p at (row_or_col)*ld + kb  where kb = (lane<16) ? 0 : 8.
static __device__ __forceinline__ v16bf load_frag16(const bf16_t* p) {
    v8bf lo = *reinterpret_cast<const v8bf*>(p);
    v8bf hi = *reinterpret_cast<const v8bf*>(p + 16);
    return __builtin_shufflevector(lo, hi, 0, 1, 2, 3, 4, 5, 6, 7,
                                           8, 9, 10, 11, 12, 13, 14, 15);
}

// ---------------- kernel 0: f32 -> bf16 convert ----------------
__global__ void mha_cvt_bf16(const float* __restrict__ src,
                             bf16_t* __restrict__ dst, int n) {
    int i = blockIdx.x * blockDim.x + threadIdx.x;
    if (i < n) dst[i] = f2bf(src[i]);
}

// ---------------- kernel 1: projection GEMM ----------------
// C[m,n] = sum_k A[m,k]*W[n,k] + bias[n];  A f32 [M,1024], W bf16 [1024,1024]
// mode 0: store qh/kh head-major   out[(hb)*S*DH + s*DH + dh]
// mode 1: store v transposed       out[(hb)*DH*S + dh*S + s]
#define TM 128
#define TN 128
#define TK 32
#define LDST 40   // padded LDS row stride (bf16 elements); 80B = 5*16B -> conflict-free

__global__ __launch_bounds__(256) void mha_gemm_proj(
    const float* __restrict__ A, const bf16_t* __restrict__ W,
    const float* __restrict__ bias, bf16_t* __restrict__ out, int mode) {
    __shared__ bf16_t sA[TM * LDST];
    __shared__ bf16_t sW[TN * LDST];

    const int t = threadIdx.x;
    const int lane = t & 31;
    const int wave = t >> 5;        // 0..7
    const int wm = wave >> 2;       // 0..1  (64 rows each)
    const int wn = wave & 3;        // 0..3  (32 cols each)
    const int mBase = blockIdx.y * TM;
    const int nBase = blockIdx.x * TN;
    const int r15 = lane & 15;
    const int kb = (lane < 16) ? 0 : 8;

    const int ldRow = t >> 1;              // 0..127
    const int ldCol = (t & 1) * 16;        // 0 or 16

    v8f acc[4][2];
#pragma unroll
    for (int i = 0; i < 4; ++i)
#pragma unroll
        for (int j = 0; j < 2; ++j) acc[i][j] = (v8f){0.f,0.f,0.f,0.f,0.f,0.f,0.f,0.f};

    for (int k0 = 0; k0 < D_; k0 += TK) {
        // ---- stage A tile (f32 -> bf16) ----
        {
            const float* src = A + (size_t)(mBase + ldRow) * D_ + k0 + ldCol;
            if (k0 + TK < D_) __builtin_prefetch(src + TK, 0, 1);
            v4f f0 = *reinterpret_cast<const v4f*>(src);
            v4f f1 = *reinterpret_cast<const v4f*>(src + 4);
            v4f f2 = *reinterpret_cast<const v4f*>(src + 8);
            v4f f3 = *reinterpret_cast<const v4f*>(src + 12);
            bf16_t* d = sA + ldRow * LDST + ldCol;
#pragma unroll
            for (int i = 0; i < 4; ++i) {
                d[i]      = f2bf(f0[i]);
                d[4 + i]  = f2bf(f1[i]);
                d[8 + i]  = f2bf(f2[i]);
                d[12 + i] = f2bf(f3[i]);
            }
        }
        // ---- stage W tile (bf16 copy) ----
        {
            const bf16_t* src = W + (size_t)(nBase + ldRow) * D_ + k0 + ldCol;
            if (k0 + TK < D_) __builtin_prefetch(src + TK, 0, 1);
            v8bf w0 = *reinterpret_cast<const v8bf*>(src);
            v8bf w1 = *reinterpret_cast<const v8bf*>(src + 8);
            *reinterpret_cast<v8bf*>(sW + ldRow * LDST + ldCol)     = w0;
            *reinterpret_cast<v8bf*>(sW + ldRow * LDST + ldCol + 8) = w1;
        }
        __syncthreads();

        v16bf bf[2];
#pragma unroll
        for (int ni = 0; ni < 2; ++ni) {
            int c = wn * 32 + ni * 16 + r15;
            bf[ni] = load_frag16(sW + c * LDST + kb);
        }
#pragma unroll
        for (int mi = 0; mi < 4; ++mi) {
            int rr = wm * 64 + mi * 16 + r15;
            v16bf af = load_frag16(sA + rr * LDST + kb);
#pragma unroll
            for (int ni = 0; ni < 2; ++ni)
                acc[mi][ni] = wmma_bf16(af, bf[ni], acc[mi][ni]);
        }
        __syncthreads();
    }

    // ---- epilogue: bias + scatter to head-major bf16 ----
    const int mOff = (lane >= 16) ? 8 : 0;
#pragma unroll
    for (int mi = 0; mi < 4; ++mi) {
#pragma unroll
        for (int ni = 0; ni < 2; ++ni) {
            int n = nBase + wn * 32 + ni * 16 + r15;
            float bv = bias[n];
            int h = n >> 6, dh = n & (DH_ - 1);
#pragma unroll
            for (int e = 0; e < 8; ++e) {
                int m = mBase + wm * 64 + mi * 16 + e + mOff;
                int b = m >> 11, s = m & (S_ - 1);
                int hb = h * B_ + b;
                size_t idx = (mode == 0)
                                 ? ((size_t)hb * S_ + s) * DH_ + dh
                                 : ((size_t)hb * DH_ + dh) * S_ + s;
                out[idx] = f2bf(acc[mi][ni][e] + bv);
            }
        }
    }
}

// ---------------- kernel 2: attention logits  Q Kt / sqrt(dh) ----------------
// grid: (S/128, S/32, HB); block 256 = 8 waves; wave tile 16(M) x 32(N)
__global__ __launch_bounds__(256) void mha_logits(
    const bf16_t* __restrict__ qh, const bf16_t* __restrict__ kh,
    float* __restrict__ attn) {
    const int head = blockIdx.z;
    const int mBase = blockIdx.y * 32;
    const int nBase = blockIdx.x * 128;
    const int lane = threadIdx.x & 31;
    const int wave = threadIdx.x >> 5;
    const int wm = wave >> 2;      // 0..1
    const int wn = wave & 3;       // 0..3
    const int r15 = lane & 15;
    const int kb = (lane < 16) ? 0 : 8;

    const bf16_t* qb = qh + (size_t)head * S_ * DH_;
    const bf16_t* kbp = kh + (size_t)head * S_ * DH_;

    const bf16_t* ap = qb + (size_t)(mBase + wm * 16 + r15) * DH_;
    v16bf a0 = load_frag16(ap + kb);        // K 0..31
    v16bf a1 = load_frag16(ap + 32 + kb);   // K 32..63

    v8f acc[2];
#pragma unroll
    for (int ni = 0; ni < 2; ++ni) acc[ni] = (v8f){0.f,0.f,0.f,0.f,0.f,0.f,0.f,0.f};

#pragma unroll
    for (int ni = 0; ni < 2; ++ni) {
        const bf16_t* bp = kbp + (size_t)(nBase + wn * 32 + ni * 16 + r15) * DH_;
        v16bf b0 = load_frag16(bp + kb);
        v16bf b1 = load_frag16(bp + 32 + kb);
        acc[ni] = wmma_bf16(a0, b0, acc[ni]);
        acc[ni] = wmma_bf16(a1, b1, acc[ni]);
    }

    float* outp = attn + (size_t)head * S_ * S_;
    const int mOff = (lane >= 16) ? 8 : 0;
#pragma unroll
    for (int ni = 0; ni < 2; ++ni) {
        int n = nBase + wn * 32 + ni * 16 + r15;
#pragma unroll
        for (int e = 0; e < 8; ++e) {
            int m = mBase + wm * 16 + e + mOff;
            outp[(size_t)m * S_ + n] = acc[ni][e] * TEMP_INV;
        }
    }
}

// ---------------- kernel 3: row softmax over 2048 (in place) ----------------
__global__ __launch_bounds__(256) void mha_softmax(float* __restrict__ attn) {
    __shared__ float red[256];
    const int t = threadIdx.x;
    float* row = attn + (size_t)blockIdx.x * S_;

    float v[8];
    float mx = -3.402823e38f;
#pragma unroll
    for (int i = 0; i < 8; ++i) {
        v[i] = row[t + i * 256];
        mx = fmaxf(mx, v[i]);
    }
    red[t] = mx; __syncthreads();
    for (int s = 128; s > 0; s >>= 1) {
        if (t < s) red[t] = fmaxf(red[t], red[t + s]);
        __syncthreads();
    }
    mx = red[0]; __syncthreads();

    float sum = 0.f;
#pragma unroll
    for (int i = 0; i < 8; ++i) {
        v[i] = __expf(v[i] - mx);
        sum += v[i];
    }
    red[t] = sum; __syncthreads();
    for (int s = 128; s > 0; s >>= 1) {
        if (t < s) red[t] += red[t + s];
        __syncthreads();
    }
    float inv = 1.0f / red[0];
#pragma unroll
    for (int i = 0; i < 8; ++i) row[t + i * 256] = v[i] * inv;
}

// ---------------- kernel 4: ctx = P @ V ----------------
// grid: (1, S/32, HB); wave tile 16(M) x 16(N); N total = DH = 64 (wn 0..3)
__global__ __launch_bounds__(256) void mha_ctx(
    const float* __restrict__ P, const bf16_t* __restrict__ vt,
    bf16_t* __restrict__ ctx) {
    const int head = blockIdx.z;              // h*B + b
    const int mBase = blockIdx.y * 32;
    const int lane = threadIdx.x & 31;
    const int wave = threadIdx.x >> 5;
    const int wm = wave >> 2;                 // 0..1
    const int wn = wave & 3;                  // 0..3
    const int r15 = lane & 15;
    const int kb = (lane < 16) ? 0 : 8;

    const float* Pp = P + (size_t)head * S_ * S_;
    const bf16_t* vp = vt + (size_t)head * DH_ * S_;

    const int m = mBase + wm * 16 + r15;      // P row this lane reads
    const int c = wn * 16 + r15;              // V column (dh) this lane reads

    v8f acc = (v8f){0.f,0.f,0.f,0.f,0.f,0.f,0.f,0.f};
    for (int k = 0; k < S_; k += 32) {
        const float* pa = Pp + (size_t)m * S_ + k + kb;
        v4f f0 = *reinterpret_cast<const v4f*>(pa);
        v4f f1 = *reinterpret_cast<const v4f*>(pa + 4);
        v4f f2 = *reinterpret_cast<const v4f*>(pa + 16);
        v4f f3 = *reinterpret_cast<const v4f*>(pa + 20);
        v16bf a;
#pragma unroll
        for (int i = 0; i < 4; ++i) {
            a[i]      = f2bf(f0[i]);
            a[4 + i]  = f2bf(f1[i]);
            a[8 + i]  = f2bf(f2[i]);
            a[12 + i] = f2bf(f3[i]);
        }
        v16bf b = load_frag16(vp + (size_t)c * S_ + k + kb);
        acc = wmma_bf16(a, b, acc);
    }

    // merge heads: ctx[b, s, h*64+dh]
    const int h = head >> 1, bidx = head & 1;
    const int mOff = (lane >= 16) ? 8 : 0;
    const int dh = wn * 16 + r15;
#pragma unroll
    for (int e = 0; e < 8; ++e) {
        int s = mBase + wm * 16 + e + mOff;
        ctx[((size_t)bidx * S_ + s) * D_ + h * DH_ + dh] = f2bf(acc[e]);
    }
}

// ---------------- kernel 5: output GEMM + bias + residual ----------------
// x[m,n] = resid[m,n] + bias[n] + sum_k ctx[m,k]*Wf[n,k]
// A tile staged via GLOBAL_LOAD_ASYNC_TO_LDS_B128 (ASYNCcnt path);
// W tile staged via a TDM descriptor + tensor_load_to_lds (TENSORcnt path),
// with pad_enable producing the conflict-free stride-40 LDS layout directly.
__global__ __launch_bounds__(256) void mha_gemm_out(
    const bf16_t* __restrict__ A, const bf16_t* __restrict__ W,
    const float* __restrict__ bias, const float* __restrict__ resid,
    float* __restrict__ xout) {
    __shared__ bf16_t sA[TM * LDST];
    __shared__ bf16_t sW[TN * LDST];

    const int t = threadIdx.x;
    const int lane = t & 31;
    const int wave = t >> 5;
    const int wm = wave >> 2;
    const int wn = wave & 3;
    const int mBase = blockIdx.y * TM;
    const int nBase = blockIdx.x * TN;
    const int r15 = lane & 15;
    const int kb = (lane < 16) ? 0 : 8;
    const int ldRow = t >> 1;
    const int ldCol = (t & 1) * 16;

    const unsigned ldsAbase = (unsigned)(size_t)(void*)sA;
    const unsigned ldsWbase = (unsigned)(size_t)(void*)sW;
    const unsigned long long Abase = (unsigned long long)(size_t)A;
    const unsigned long long Wbase = (unsigned long long)(size_t)W;

    v8f acc[4][2];
#pragma unroll
    for (int i = 0; i < 4; ++i)
#pragma unroll
        for (int j = 0; j < 2; ++j) acc[i][j] = (v8f){0.f,0.f,0.f,0.f,0.f,0.f,0.f,0.f};

    for (int k0 = 0; k0 < D_; k0 += TK) {
        // ---- A tile: async copy global -> LDS, bypassing VGPRs ----
        {
            unsigned gOff = (unsigned)(((mBase + ldRow) * D_ + k0 + ldCol) * 2);
            unsigned lOff = ldsAbase + (unsigned)((ldRow * LDST + ldCol) * 2);
            asm volatile("global_load_async_to_lds_b128 %0, %1, %2"
                         :: "v"(lOff), "v"(gOff), "s"(Abase) : "memory");
            asm volatile("global_load_async_to_lds_b128 %0, %1, %2"
                         :: "v"(lOff + 16u), "v"(gOff + 16u), "s"(Abase) : "memory");
        }
        // ---- W tile: one TDM 2D tensor load per K-step, issued by wave 0 ----
        if (t < 32) {
            unsigned long long ga =
                Wbase + (unsigned long long)((size_t)nBase * D_ + (size_t)k0) * 2ull;
            uint4v g0;
            g0.x = 1u;                                   // count=1, user mode
            g0.y = ldsWbase;                             // lds_addr (bytes)
            g0.z = (unsigned)(ga & 0xFFFFFFFFull);       // global_addr[31:0]
            g0.w = (unsigned)((ga >> 32) & 0x01FFFFFFull) | 0x80000000u; // [56:32] | type=2
            uint8v g1;
            // data_size=2B (code 1), pad_enable, pad_interval=16 DW (code 3),
            // pad_amount=4 DW (code 3) -> LDS row stride = 64B + 16B = LDST*2
            g1.s0 = (1u << 16) | (1u << 20) | (3u << 22) | (3u << 25);
            g1.s1 = (unsigned)(D_ & 0xFFFF) << 16;       // tensor_dim0[15:0]=1024
            g1.s2 = ((unsigned)(D_ >> 16) & 0xFFFFu) |   // tensor_dim0[31:16]
                    ((unsigned)(D_ & 0xFFFF) << 16);     // tensor_dim1[15:0]=1024
            g1.s3 = ((unsigned)(D_ >> 16) & 0xFFFFu) |   // tensor_dim1[31:16]
                    ((unsigned)TK << 16);                // tile_dim0 = 32 elements
            g1.s4 = (unsigned)TN;                        // tile_dim1 = 128 rows; tile_dim2=0
            g1.s5 = (unsigned)D_;                        // tensor_dim0_stride = 1024 (lo32)
            g1.s6 = 0u;                                  // stride0 hi16, stride1 lo16
            g1.s7 = 0u;                                  // stride1 hi32
            asm volatile("tensor_load_to_lds %0, %1" :: "s"(g0), "s"(g1) : "memory");
        }
        __builtin_amdgcn_s_wait_tensorcnt(0);
        asm volatile("s_wait_asynccnt 0x0" ::: "memory");
        __syncthreads();

        v16bf bf[2];
#pragma unroll
        for (int ni = 0; ni < 2; ++ni) {
            int c = wn * 32 + ni * 16 + r15;
            bf[ni] = load_frag16(sW + c * LDST + kb);
        }
#pragma unroll
        for (int mi = 0; mi < 4; ++mi) {
            int rr = wm * 64 + mi * 16 + r15;
            v16bf af = load_frag16(sA + rr * LDST + kb);
#pragma unroll
            for (int ni = 0; ni < 2; ++ni)
                acc[mi][ni] = wmma_bf16(af, bf[ni], acc[mi][ni]);
        }
        __syncthreads();
    }

    const int mOff = (lane >= 16) ? 8 : 0;
#pragma unroll
    for (int mi = 0; mi < 4; ++mi) {
#pragma unroll
        for (int ni = 0; ni < 2; ++ni) {
            int n = nBase + wn * 32 + ni * 16 + r15;
            float bv = bias[n];
#pragma unroll
            for (int e = 0; e < 8; ++e) {
                int m = mBase + wm * 64 + mi * 16 + e + mOff;
                size_t idx = (size_t)m * D_ + n;
                xout[idx] = acc[mi][ni][e] + bv + resid[idx];
            }
        }
    }
}

// ---------------- kernel 6: layernorm over D=1024 ----------------
__global__ __launch_bounds__(256) void mha_layernorm(
    const float* __restrict__ x, const float* __restrict__ gamma,
    const float* __restrict__ beta, float* __restrict__ out) {
    __shared__ float red[256];
    const int t = threadIdx.x;
    const float* xr = x + (size_t)blockIdx.x * D_;
    float* orow = out + (size_t)blockIdx.x * D_;

    float v[4];
    float s = 0.f;
#pragma unroll
    for (int i = 0; i < 4; ++i) {
        v[i] = xr[t + i * 256];
        s += v[i];
    }
    red[t] = s; __syncthreads();
    for (int k = 128; k > 0; k >>= 1) {
        if (t < k) red[t] += red[t + k];
        __syncthreads();
    }
    float mu = red[0] * (1.0f / D_); __syncthreads();

    float s2 = 0.f;
#pragma unroll
    for (int i = 0; i < 4; ++i) {
        float d = v[i] - mu;
        s2 += d * d;
    }
    red[t] = s2; __syncthreads();
    for (int k = 128; k > 0; k >>= 1) {
        if (t < k) red[t] += red[t + k];
        __syncthreads();
    }
    float rinv = rsqrtf(red[0] * (1.0f / D_) + EPS_);
#pragma unroll
    for (int i = 0; i < 4; ++i) {
        int n = t + i * 256;
        orow[n] = (v[i] - mu) * rinv * gamma[n] + beta[n];
    }
}

// ---------------- host orchestration ----------------
extern "C" void kernel_launch(void* const* d_in, const int* in_sizes, int n_in,
                              void* d_out, int out_size, void* d_ws, size_t ws_size,
                              hipStream_t stream) {
    (void)in_sizes; (void)n_in; (void)out_size; (void)ws_size;
    const float* keyp   = (const float*)d_in[0];
    const float* valp   = (const float*)d_in[1];
    const float* qryp   = (const float*)d_in[2];
    const float* Wq     = (const float*)d_in[3];
    const float* bq     = (const float*)d_in[4];
    const float* Wk     = (const float*)d_in[5];
    const float* bk     = (const float*)d_in[6];
    const float* Wv     = (const float*)d_in[7];
    const float* bv     = (const float*)d_in[8];
    const float* Wf     = (const float*)d_in[9];
    const float* bf_    = (const float*)d_in[10];
    const float* gamma  = (const float*)d_in[11];
    const float* beta   = (const float*)d_in[12];

    float* outF = (float*)d_out;
    float* attn = outF + (size_t)M_ * D_;       // [32, 2048, 2048] f32

    // workspace carve-up (bf16/f32), total ~56 MB
    bf16_t* Wqb  = (bf16_t*)d_ws;
    bf16_t* Wkb  = Wqb + (size_t)D_ * D_;
    bf16_t* Wvb  = Wkb + (size_t)D_ * D_;
    bf16_t* Wfb  = Wvb + (size_t)D_ * D_;
    bf16_t* qh   = Wfb + (size_t)D_ * D_;       // [HB, S, DH]
    bf16_t* kh   = qh  + (size_t)HB_ * S_ * DH_;
    bf16_t* vt   = kh  + (size_t)HB_ * S_ * DH_;   // [HB, DH, S]
    bf16_t* ctxb = vt  + (size_t)HB_ * S_ * DH_;   // [B, S, D]
    float*  xbuf = (float*)(ctxb + (size_t)M_ * D_);

    const int nw = D_ * D_;
    mha_cvt_bf16<<<(nw + 255) / 256, 256, 0, stream>>>(Wq, Wqb, nw);
    mha_cvt_bf16<<<(nw + 255) / 256, 256, 0, stream>>>(Wk, Wkb, nw);
    mha_cvt_bf16<<<(nw + 255) / 256, 256, 0, stream>>>(Wv, Wvb, nw);
    mha_cvt_bf16<<<(nw + 255) / 256, 256, 0, stream>>>(Wf, Wfb, nw);

    dim3 gGemm(D_ / TN, M_ / TM);               // (8, 32)
    mha_gemm_proj<<<gGemm, 256, 0, stream>>>(qryp, Wqb, bq, qh, 0);
    mha_gemm_proj<<<gGemm, 256, 0, stream>>>(keyp, Wkb, bk, kh, 0);
    mha_gemm_proj<<<gGemm, 256, 0, stream>>>(valp, Wvb, bv, vt, 1);

    mha_logits<<<dim3(S_ / 128, S_ / 32, HB_), 256, 0, stream>>>(qh, kh, attn);
    mha_softmax<<<HB_ * S_, 256, 0, stream>>>(attn);
    mha_ctx<<<dim3(1, S_ / 32, HB_), 256, 0, stream>>>(attn, vt, ctxb);

    mha_gemm_out<<<gGemm, 256, 0, stream>>>(ctxb, Wfb, bf_, qryp, xbuf);
    mha_layernorm<<<M_, 256, 0, stream>>>(xbuf, gamma, beta, outF);
}
